// GlobalSceneTokenizer_11407433138453
// MI455X (gfx1250) — matrix operand order repaired and checked by reading, not compiled, not run on hardware
//
#include <hip/hip_runtime.h>
#include <hip/hip_bf16.h>

typedef __attribute__((ext_vector_type(16))) _Float16 v16h;
typedef __attribute__((ext_vector_type(8)))  _Float16 v8h;
typedef __attribute__((ext_vector_type(8)))  float    v8f;

#define BATCH 4
#define NPTS  4096
#define KNBR  32
#define CSTEM 64
#define CIN   187
#define CINP  192
#define CTOK  256
#define RAD2  (0.16f*0.16f)
#define LOG1E4 9.210340371976184f

// ---------------------------------------------------------------------------
// Kernel 1: weight fragmentation. Lay out W1 (padded 187->192) and W2 as f16
// in per-lane WMMA B-fragment order: frag[((ks*16+nt)*32+lane)*16 + j]
// where k = ks*32 + (lane>>4)*16 + j, ncol = nt*16 + (lane&15).
// ---------------------------------------------------------------------------
__global__ void prep_kernel(const float* __restrict__ w1,
                            const float* __restrict__ w2,
                            _Float16* __restrict__ f1,
                            _Float16* __restrict__ f2) {
  int gid = blockIdx.x * 256 + threadIdx.x;
  const int N1 = 6 * 16 * 32 * 16;   // 49152
  const int N2 = 8 * 16 * 32 * 16;   // 65536
  if (gid < N1) {
    int j = gid & 15, lane = (gid >> 4) & 31, nt = (gid >> 9) & 15, ks = gid >> 13;
    int k = ks * 32 + (lane >> 4) * 16 + j;
    int ncol = nt * 16 + (lane & 15);
    float v = (k < CIN) ? w1[k * CTOK + ncol] : 0.f;
    f1[gid] = (_Float16)v;
  } else {
    int g = gid - N1;
    if (g < N2) {
      int j = g & 15, lane = (g >> 4) & 31, nt = (g >> 9) & 15, ks = g >> 13;
      int k = ks * 32 + (lane >> 4) * 16 + j;
      int ncol = nt * 16 + (lane & 15);
      f2[g] = (_Float16)w2[k * CTOK + ncol];
    }
  }
}

// ---------------------------------------------------------------------------
// Kernel 2: ball query. One thread per query point; LDS-tiled scan over all
// N points; collect first 32 in-ball indices (ascending j), pad with first.
// ---------------------------------------------------------------------------
__global__ void bq_kernel(const float* __restrict__ xyz, int* __restrict__ idxb) {
  __shared__ float tx[256], ty[256], tz[256];
  int tid = threadIdx.x;
  int b = blockIdx.x >> 4;               // 16 blocks per batch
  int n = ((blockIdx.x & 15) << 8) + tid;
  const float* xb = xyz + (size_t)b * NPTS * 3;
  float qx = xb[n * 3], qy = xb[n * 3 + 1], qz = xb[n * 3 + 2];
  int* myidx = idxb + (size_t)(b * NPTS + n) * KNBR;
  int cnt = 0;
  for (int jt = 0; jt < NPTS; jt += 256) {
    int j = jt + tid;
    tx[tid] = xb[j * 3]; ty[tid] = xb[j * 3 + 1]; tz[tid] = xb[j * 3 + 2];
    __syncthreads();
    if (cnt < KNBR) {
      for (int jj = 0; jj < 256 && cnt < KNBR; ++jj) {
        float dx = tx[jj] - qx, dy = ty[jj] - qy, dz = tz[jj] - qz;
        float sq = dx * dx + dy * dy + dz * dz;
        if (sq <= RAD2) myidx[cnt++] = jt + jj;
      }
    }
    __syncthreads();
  }
  int first = (cnt > 0) ? myidx[0] : n;
  for (int r = cnt; r < KNBR; ++r) myidx[r] = first;
}

// ---------------------------------------------------------------------------
// WMMA fragment loaders.
// A (16-bit, 16x32, row-major source): lane halves hold K {0-15} /
// {16-31} -> two contiguous 8-f16 runs per lane.
// ---------------------------------------------------------------------------
__device__ __forceinline__ v16h load_a_frag(const _Float16* row0, int stride,
                                            int mtile, int ks, int lane) {
  int m = mtile * 16 + (lane & 15);
  int kh = lane >> 4;
  const _Float16* p = row0 + m * stride + ks * 32 + kh * 16;
  v8h lo = *(const v8h*)p;
  v8h hi = *(const v8h*)(p + 8);
  return __builtin_shufflevector(lo, hi, 0, 1, 2, 3, 4, 5, 6, 7,
                                 8, 9, 10, 11, 12, 13, 14, 15);
}

__device__ __forceinline__ v16h load_b_frag(const _Float16* __restrict__ frag,
                                            int ks, int nt, int lane) {
  return *(const v16h*)(frag + (((ks * 16 + nt) * 32 + lane) << 4));
}

// ---------------------------------------------------------------------------
// Kernel 3: fused grouped-MLP. One block (8 waves) per (b,n):
//  build grouped input [32x192] f16 in LDS -> WMMA layer1 -> bias+LN+GELU ->
//  WMMA layer2 -> bias+LN fused with max over K -> sceneFeat[b,n,256].
// ---------------------------------------------------------------------------
__global__ void __launch_bounds__(256) main_kernel(
    const float* __restrict__ xyz, const float* __restrict__ feat,
    const int* __restrict__ idxb,
    const float* __restrict__ b1, const float* __restrict__ g1, const float* __restrict__ be1,
    const float* __restrict__ b2, const float* __restrict__ g2, const float* __restrict__ be2,
    const _Float16* __restrict__ fragW1, const _Float16* __restrict__ fragW2,
    float* __restrict__ sceneFeat) {
  __shared__ _Float16 A1[KNBR][CINP];   // 12 KB
  __shared__ float    Hs[KNBR][CTOK];   // 32 KB
  __shared__ _Float16 A2[KNBR][CTOK];   // 16 KB
  __shared__ float absemb[96];
  __shared__ float relL[KNBR][3];
  __shared__ int   nbrL[KNBR];
  __shared__ float muL[KNBR], rstdL[KNBR];

  int tid = threadIdx.x;
  int lane = tid & 31, w = tid >> 5;
  int b = blockIdx.x >> 12, n = blockIdx.x & 4095;
  const float* center = xyz + (size_t)(b * NPTS + n) * 3;
  float cx = center[0], cy = center[1], cz = center[2];

  // absolute position embedding (96) for the center, shared by all K rows
  if (tid < 96) {
    int axis = tid >> 5, k = tid & 31;
    float cc = (axis == 0) ? cx : ((axis == 1) ? cy : cz);
    float fr = __expf(-LOG1E4 * (float)(k & 15) * (1.f / 15.f));
    float ang = cc * fr;
    absemb[tid] = (k < 16) ? __sinf(ang) : __cosf(ang);
  }
  if (tid < KNBR) {
    int j = idxb[(size_t)(b * NPTS + n) * KNBR + tid];
    nbrL[tid] = j;
    const float* pj = xyz + (size_t)(b * NPTS + j) * 3;
    relL[tid][0] = pj[0] - cx; relL[tid][1] = pj[1] - cy; relL[tid][2] = pj[2] - cz;
  }
  __syncthreads();

  // build grouped input [32 x 192] (f16): feat(64) | rel(3) | relemb(24) | absemb(96) | pad
  for (int e = tid; e < KNBR * CINP; e += 256) {
    int r = e / CINP, c = e - r * CINP;
    float v;
    if (c < 64) {
      v = feat[(size_t)(b * NPTS + nbrL[r]) * CSTEM + c];
    } else if (c < 67) {
      v = relL[r][c - 64];
    } else if (c < 91) {
      int cc = c - 67, axis = cc >> 3, k = cc & 7;
      float fr = __expf(-LOG1E4 * (float)(k & 3) * (1.f / 3.f));
      float ang = relL[r][axis] * fr;
      v = (k < 4) ? __sinf(ang) : __cosf(ang);
    } else if (c < 187) {
      v = absemb[c - 91];
    } else {
      v = 0.f;
    }
    A1[r][c] = (_Float16)v;
  }
  __syncthreads();

  // ---- layer 1: [32x192] @ [192x256]  (6 K-steps of 32) ----
  int nt0 = w * 2, nt1 = nt0 + 1;
  v8f c00 = {}, c01 = {}, c10 = {}, c11 = {};
  for (int ks = 0; ks < 6; ++ks) {
    v16h af0 = load_a_frag(&A1[0][0], CINP, 0, ks, lane);
    v16h af1 = load_a_frag(&A1[0][0], CINP, 1, ks, lane);
    v16h bf0 = load_b_frag(fragW1, ks, nt0, lane);
    v16h bf1 = load_b_frag(fragW1, ks, nt1, lane);
    c00 = __builtin_amdgcn_wmma_f32_16x16x32_f16(false, af0, false, bf0, (short)0, c00, false, false);
    c01 = __builtin_amdgcn_wmma_f32_16x16x32_f16(false, af0, false, bf1, (short)0, c01, false, false);
    c10 = __builtin_amdgcn_wmma_f32_16x16x32_f16(false, af1, false, bf0, (short)0, c10, false, false);
    c11 = __builtin_amdgcn_wmma_f32_16x16x32_f16(false, af1, false, bf1, (short)0, c11, false, false);
  }
  {
    int colr = lane & 15, half = lane >> 4;
    float bias0 = b1[nt0 * 16 + colr], bias1 = b1[nt1 * 16 + colr];
    for (int v = 0; v < 8; ++v) {
      int m = v + half * 8;  // C layout: VGPR v, lane-half selects M / M+8
      Hs[m][nt0 * 16 + colr]      = c00[v] + bias0;
      Hs[m][nt1 * 16 + colr]      = c01[v] + bias1;
      Hs[16 + m][nt0 * 16 + colr] = c10[v] + bias0;
      Hs[16 + m][nt1 * 16 + colr] = c11[v] + bias1;
    }
  }
  __syncthreads();
  if (tid < KNBR) {
    float s = 0.f, s2 = 0.f;
    for (int c = 0; c < CTOK; ++c) { float x = Hs[tid][c]; s += x; s2 += x * x; }
    float mu = s * (1.f / CTOK);
    float var = s2 * (1.f / CTOK) - mu * mu;
    muL[tid] = mu; rstdL[tid] = rsqrtf(var + 1e-5f);
  }
  __syncthreads();
  for (int e = tid; e < KNBR * CTOK; e += 256) {
    int r = e >> 8, c = e & 255;
    float x = (Hs[r][c] - muL[r]) * rstdL[r] * g1[c] + be1[c];
    float gel = 0.5f * x * (1.f + erff(x * 0.70710678118654752f));
    A2[r][c] = (_Float16)gel;
  }
  __syncthreads();

  // ---- layer 2: [32x256] @ [256x256]  (8 K-steps of 32) ----
  c00 = (v8f){}; c01 = (v8f){}; c10 = (v8f){}; c11 = (v8f){};
  for (int ks = 0; ks < 8; ++ks) {
    v16h af0 = load_a_frag(&A2[0][0], CTOK, 0, ks, lane);
    v16h af1 = load_a_frag(&A2[0][0], CTOK, 1, ks, lane);
    v16h bf0 = load_b_frag(fragW2, ks, nt0, lane);
    v16h bf1 = load_b_frag(fragW2, ks, nt1, lane);
    c00 = __builtin_amdgcn_wmma_f32_16x16x32_f16(false, af0, false, bf0, (short)0, c00, false, false);
    c01 = __builtin_amdgcn_wmma_f32_16x16x32_f16(false, af0, false, bf1, (short)0, c01, false, false);
    c10 = __builtin_amdgcn_wmma_f32_16x16x32_f16(false, af1, false, bf0, (short)0, c10, false, false);
    c11 = __builtin_amdgcn_wmma_f32_16x16x32_f16(false, af1, false, bf1, (short)0, c11, false, false);
  }
  {
    int colr = lane & 15, half = lane >> 4;
    float bias0 = b2[nt0 * 16 + colr], bias1 = b2[nt1 * 16 + colr];
    for (int v = 0; v < 8; ++v) {
      int m = v + half * 8;
      Hs[m][nt0 * 16 + colr]      = c00[v] + bias0;
      Hs[m][nt1 * 16 + colr]      = c01[v] + bias1;
      Hs[16 + m][nt0 * 16 + colr] = c10[v] + bias0;
      Hs[16 + m][nt1 * 16 + colr] = c11[v] + bias1;
    }
  }
  __syncthreads();
  if (tid < KNBR) {
    float s = 0.f, s2 = 0.f;
    for (int c = 0; c < CTOK; ++c) { float x = Hs[tid][c]; s += x; s2 += x * x; }
    float mu = s * (1.f / CTOK);
    float var = s2 * (1.f / CTOK) - mu * mu;
    muL[tid] = mu; rstdL[tid] = rsqrtf(var + 1e-5f);
  }
  __syncthreads();
  {
    int c = tid;
    float gg = g2[c], bb = be2[c];
    float mx = -3.4e38f;
    for (int r = 0; r < KNBR; ++r) {
      float x = (Hs[r][c] - muL[r]) * rstdL[r] * gg + bb;
      mx = fmaxf(mx, x);
    }
    sceneFeat[(size_t)(b * NPTS + n) * CTOK + c] = mx;
  }
}

// ---------------------------------------------------------------------------
// Kernel 4: partial max over N (16 chunks of 256 rows per batch).
// ---------------------------------------------------------------------------
__global__ void pmax_kernel(const float* __restrict__ sceneFeat,
                            float* __restrict__ pmaxb) {
  int b = blockIdx.x >> 4, ch = blockIdx.x & 15, c = threadIdx.x;
  int nb = ch << 8;
  float mx = -3.4e38f;
  for (int i = 0; i < 256; ++i)
    mx = fmaxf(mx, sceneFeat[(size_t)(b * NPTS + nb + i) * CTOK + c]);
  pmaxb[(size_t)blockIdx.x * CTOK + c] = mx;
}

// ---------------------------------------------------------------------------
// Kernel 5: final pooling + two small 256x256 MLPs. One block per batch.
// ---------------------------------------------------------------------------
__device__ __forceinline__ float block_sum(float v, float* red, int tid) {
  red[tid] = v; __syncthreads();
  for (int s = 128; s > 0; s >>= 1) {
    if (tid < s) red[tid] += red[tid + s];
    __syncthreads();
  }
  float r = red[0]; __syncthreads();
  return r;
}

__global__ void final_kernel(const float* __restrict__ pmaxb,
    const float* __restrict__ pw1, const float* __restrict__ pb1,
    const float* __restrict__ pg1, const float* __restrict__ pbe1,
    const float* __restrict__ pw2, const float* __restrict__ pb2,
    const float* __restrict__ pg2, const float* __restrict__ pbe2,
    float* __restrict__ out) {
  __shared__ float vecA[CTOK], vecB[CTOK], red[CTOK];
  int b = blockIdx.x, c = threadIdx.x;
  float mx = -3.4e38f;
  for (int ch = 0; ch < 16; ++ch)
    mx = fmaxf(mx, pmaxb[(size_t)(b * 16 + ch) * CTOK + c]);
  vecA[c] = mx; __syncthreads();

  float acc = pb1[c];
  for (int i = 0; i < CTOK; ++i) acc += vecA[i] * pw1[i * CTOK + c];
  float s  = block_sum(acc, red, c);
  float s2 = block_sum(acc * acc, red, c);
  float mu = s * (1.f / CTOK);
  float rstd = rsqrtf(s2 * (1.f / CTOK) - mu * mu + 1e-5f);
  float x = (acc - mu) * rstd * pg1[c] + pbe1[c];
  float gel = 0.5f * x * (1.f + erff(x * 0.70710678118654752f));
  vecB[c] = gel; __syncthreads();

  float acc2 = pb2[c];
  for (int i = 0; i < CTOK; ++i) acc2 += vecB[i] * pw2[i * CTOK + c];
  float t  = block_sum(acc2, red, c);
  float t2 = block_sum(acc2 * acc2, red, c);
  float mu2 = t * (1.f / CTOK);
  float rstd2 = rsqrtf(t2 * (1.f / CTOK) - mu2 * mu2 + 1e-5f);
  out[(size_t)b * CTOK + c] = (acc2 - mu2) * rstd2 * pg2[c] + pbe2[c];
}

// ---------------------------------------------------------------------------
// Launch
// ---------------------------------------------------------------------------
extern "C" void kernel_launch(void* const* d_in, const int* in_sizes, int n_in,
                              void* d_out, int out_size, void* d_ws, size_t ws_size,
                              hipStream_t stream) {
  const float* xyz  = (const float*)d_in[0];
  const float* feat = (const float*)d_in[1];
  const float* w1   = (const float*)d_in[2];
  const float* b1   = (const float*)d_in[3];
  const float* g1   = (const float*)d_in[4];
  const float* be1  = (const float*)d_in[5];
  const float* w2   = (const float*)d_in[6];
  const float* b2   = (const float*)d_in[7];
  const float* g2   = (const float*)d_in[8];
  const float* be2  = (const float*)d_in[9];
  const float* pw1  = (const float*)d_in[10];
  const float* pb1  = (const float*)d_in[11];
  const float* pg1  = (const float*)d_in[12];
  const float* pbe1 = (const float*)d_in[13];
  const float* pw2  = (const float*)d_in[14];
  const float* pb2  = (const float*)d_in[15];
  const float* pg2  = (const float*)d_in[16];
  const float* pbe2 = (const float*)d_in[17];
  float* out = (float*)d_out;

  char* ws = (char*)d_ws;
  _Float16* fragW1 = (_Float16*)(ws);                 //  96 KB
  _Float16* fragW2 = (_Float16*)(ws + 98304);         // 128 KB
  int*   idxb      = (int*)  (ws + 229376);           //   2 MB
  float* sceneFeat = (float*)(ws + 2326528);          //  16 MB
  float* pmaxBuf   = (float*)(ws + 19103744);         //  64 KB

  hipLaunchKernelGGL(prep_kernel, dim3(448), dim3(256), 0, stream, w1, w2, fragW1, fragW2);
  hipLaunchKernelGGL(bq_kernel, dim3(64), dim3(256), 0, stream, xyz, idxb);
  hipLaunchKernelGGL(main_kernel, dim3(BATCH * NPTS), dim3(256), 0, stream,
                     xyz, feat, idxb, b1, g1, be1, b2, g2, be2, fragW1, fragW2, sceneFeat);
  hipLaunchKernelGGL(pmax_kernel, dim3(BATCH * 16), dim3(256), 0, stream, sceneFeat, pmaxBuf);
  hipLaunchKernelGGL(final_kernel, dim3(BATCH), dim3(256), 0, stream, pmaxBuf,
                     pw1, pb1, pg1, pbe1, pw2, pb2, pg2, pbe2, out);
}
